// SugarPairformerLayer_22101901705609
// MI455X (gfx1250) — matrix operand order repaired and checked by reading, not compiled, not run on hardware
//
#include <hip/hip_runtime.h>
#include <hip/hip_bf16.h>

// ===========================================================================
// Pairformer layer for MI455X (gfx1250, wave32, WMMA).
// All matrix contractions run through v_wmma_f32_16x16x32_f16 (f16 in, fp32
// accum). LN / softmax / gates / residuals stay fp32. z (33.5MB) is L2
// resident; the layer is compute-bound on WMMA, so operand staging uses the
// CDNA5 async paths: global_load_async_to_lds_b128 (ASYNCcnt) in the GEMM and
// the Tensor Data Mover (TENSORcnt) in the triangle einsum.
// ===========================================================================

typedef __attribute__((ext_vector_type(16))) _Float16 hv16;
typedef __attribute__((ext_vector_type(8)))  _Float16 hv8;
typedef __attribute__((ext_vector_type(8)))  float    fv8;
typedef __attribute__((ext_vector_type(4)))  unsigned int u32x4;
typedef __attribute__((ext_vector_type(8)))  int      i32x8;
typedef __attribute__((ext_vector_type(4)))  int      i32x4;

#define NTOK 256
#define NN   65536      // 256*256
#define DZ   128
#define CS   384
#define HT   4          // tri heads
#define WT   32         // tri head width
#define HP   16         // pair heads
#define DP   24         // pair head dim
#define INFB 1e9f

#ifndef __has_builtin
#define __has_builtin(x) 0
#endif
#if __has_builtin(__builtin_amdgcn_tensor_load_to_lds)
#define HAVE_TDM 1
#else
#define HAVE_TDM 0
#endif

__device__ __forceinline__ float sigf(float x) { return 1.0f / (1.0f + __expf(-x)); }
__device__ __forceinline__ float siluf(float x) { return x * sigf(x); }

// ---------------------------------------------------------------------------
// CDNA5 async global->LDS (ASYNCcnt) + waits
// ---------------------------------------------------------------------------
__device__ __forceinline__ void async_g2l_b128(_Float16* lds, const _Float16* g) {
  unsigned lo = (unsigned)(uintptr_t)lds;  // low 32 bits of generic = LDS offset
  asm volatile("global_load_async_to_lds_b128 %0, %1, off"
               :: "v"(lo), "v"(g) : "memory");
}

template <int N>
__device__ __forceinline__ void wait_asynccnt() {
#if __has_builtin(__builtin_amdgcn_s_wait_asynccnt)
  __builtin_amdgcn_s_wait_asynccnt(N);
#else
  asm volatile("s_wait_asynccnt %0" :: "n"(N) : "memory");
#endif
}

__device__ __forceinline__ void wait_tensorcnt0() {
#if __has_builtin(__builtin_amdgcn_s_wait_tensorcnt)
  __builtin_amdgcn_s_wait_tensorcnt(0);
#else
  asm volatile("s_wait_tensorcnt 0x0" ::: "memory");
#endif
}

#if HAVE_TDM
// Issue a TDM 2D tile load: rows x rowHalves f16 elements, contiguous rows
// (stride == rowHalves), global -> LDS. D# packing per ISA 8.3/8.4.
// clang-23 6-arg form: (u32x4 g0, i32x8 g1, i32x4 g2, i32x4 g3, i32x8 gx, i32 cpol)
__device__ __forceinline__ void tdm_load_2d(void* lds, const void* gsrc,
                                            unsigned rows, unsigned rowHalves) {
  unsigned long long ga = (unsigned long long)(uintptr_t)gsrc;
  unsigned lo = (unsigned)(uintptr_t)lds;
  u32x4 g0;
  g0[0] = 1u;                                            // count=1, user mode
  g0[1] = lo;                                            // lds_addr
  g0[2] = (unsigned)(ga & 0xffffffffu);                  // global_addr lo
  g0[3] = (unsigned)((ga >> 32) & 0x1ffffffu) | (2u << 30);  // addr hi | type=2
  i32x8 g1;
  g1[0] = (int)(1u << 16);                               // data_size = 2 bytes
  g1[1] = (int)(rowHalves << 16);                        // tensor_dim0[15:0]
  g1[2] = (int)((rowHalves >> 16) | (rows << 16));       // dim0 hi | dim1 lo
  g1[3] = (int)((rows >> 16) | (rowHalves << 16));       // dim1 hi | tile_dim0
  g1[4] = (int)(rows & 0xffffu);                         // tile_dim1, tile_dim2=0
  g1[5] = (int)rowHalves;                                // tensor_dim0_stride lo
  g1[6] = 0;
  g1[7] = 0;
  i32x4 zz4 = (i32x4){0, 0, 0, 0};
  i32x8 zz8 = (i32x8){0, 0, 0, 0, 0, 0, 0, 0};
  __builtin_amdgcn_tensor_load_to_lds(g0, g1, zz4, zz4, zz8, 0);
}
#endif

// ---------------------------------------------------------------------------
// WMMA fragment loaders (ISA 7.12.2 layouts, wave32).
// A 16x32 f16: lane l holds row (l&15); halves 0..7 = K sel*8.., halves
//              8..15 = K 16+sel*8.., sel = l>>4.
// B 32x16 f16 (source stored transposed, [n][k]): lane l holds col (l&15),
//              same K order as A: halves 0..7 = K sel*8.., 8..15 = K 16+sel*8..
// ---------------------------------------------------------------------------
__device__ __forceinline__ hv16 load_a_frag(const _Float16* A, int lda) {
  int lane = threadIdx.x & 31;
  const _Float16* p = A + (size_t)(lane & 15) * lda + (lane >> 4) * 8;
  hv8 lo = *(const hv8*)p;
  hv8 hi = *(const hv8*)(p + 16);
  hv16 r;
#pragma unroll
  for (int t = 0; t < 8; ++t) { r[t] = lo[t]; r[8 + t] = hi[t]; }
  return r;
}

__device__ __forceinline__ hv16 load_b_frag(const _Float16* Bt, int ldb) {
  int lane = threadIdx.x & 31;
  const _Float16* p = Bt + (size_t)(lane & 15) * ldb + (lane >> 4) * 8;
  hv8 a0 = *(const hv8*)p;
  hv8 a1 = *(const hv8*)(p + 16);
  hv16 r;
#pragma unroll
  for (int t = 0; t < 8; ++t) { r[t] = a0[t]; r[8 + t] = a1[t]; }
  return r;
}

__device__ __forceinline__ fv8 wmma_f16(hv16 a, hv16 b, fv8 c) {
  return __builtin_amdgcn_wmma_f32_16x16x32_f16(false, a, false, b, (short)0, c,
                                                false, false);
}

// ---------------------------------------------------------------------------
// Elementwise helpers
// ---------------------------------------------------------------------------
__global__ void k_copy_f32(const float* __restrict__ src, float* __restrict__ dst, int n) {
  int i = blockIdx.x * 256 + threadIdx.x;
  if (i < n) dst[i] = src[i];
}

__global__ void k_pm(const float* __restrict__ mask, float* __restrict__ pm) {
  int i = blockIdx.x, j = threadIdx.x;
  pm[i * NTOK + j] = mask[i] * mask[j];
}

// weight convert: in fp32 [K][N] -> out f16 [N][K] (transposed for B frags)
__global__ void k_wt(const float* __restrict__ w, _Float16* __restrict__ out, int K, int N) {
  int idx = blockIdx.x * 256 + threadIdx.x;
  if (idx >= K * N) return;
  int n = idx / K, k = idx % K;
  out[idx] = (_Float16)w[(size_t)k * N + n];
}

// ---------------------------------------------------------------------------
// Fused LayerNorm -> f16 rows. transN>0: read source row transposed (i,j)->(j,i)
// ---------------------------------------------------------------------------
__global__ void k_ln(const float* __restrict__ x, const float* __restrict__ g,
                     const float* __restrict__ b, int D, int transN,
                     _Float16* __restrict__ outh) {
  int row = blockIdx.x;
  int srow = (transN > 0) ? ((row % transN) * transN + row / transN) : row;
  const float* xr = x + (size_t)srow * D;
  __shared__ float red[128];
  float v[3];
  int tid = threadIdx.x;
  float s = 0.f;
  int e = 0;
  for (int c = tid; c < D; c += 128, ++e) { v[e] = xr[c]; s += v[e]; }
  red[tid] = s; __syncthreads();
  for (int o = 64; o > 0; o >>= 1) { if (tid < o) red[tid] += red[tid + o]; __syncthreads(); }
  float mu = red[0] / (float)D;
  __syncthreads();
  s = 0.f; e = 0;
  for (int c = tid; c < D; c += 128, ++e) { float d = v[e] - mu; s += d * d; }
  red[tid] = s; __syncthreads();
  for (int o = 64; o > 0; o >>= 1) { if (tid < o) red[tid] += red[tid + o]; __syncthreads(); }
  float rs = rsqrtf(red[0] / (float)D + 1e-5f);
  e = 0;
  for (int c = tid; c < D; c += 128, ++e)
    outh[(size_t)row * D + c] = (_Float16)((v[e] - mu) * rs * g[c] + b[c]);
}

// ---------------------------------------------------------------------------
// Generic WMMA GEMM: out = epi(A[M][K](f16) x Bt[N][K](f16) + bias)
// 64x64 tile / block, 4 waves, double-buffered async global->LDS staging.
// Epilogues:
//  0: outf = acc                          1: outh = f16(acc)
//  3: outh = f16(silu(auxh)*acc)          4: outf[mT] += acc
//  5: outf[mT] += aux*sigmoid(acc)        7: tri-mul gate*mask -> channel-major
// ---------------------------------------------------------------------------
template <int EPI>
__global__ __launch_bounds__(128) void k_gemm(
    const _Float16* __restrict__ A, const _Float16* __restrict__ Bt, int M,
    int Nn, int K, const float* __restrict__ bias, const float* __restrict__ aux,
    const _Float16* __restrict__ auxh, float* __restrict__ outf,
    _Float16* __restrict__ outh, const float* __restrict__ pm,
    _Float16* __restrict__ scat, int scat_trans, int resTransN) {
  __shared__ __align__(16) _Float16 sA[2][64 * 32];
  __shared__ __align__(16) _Float16 sB[2][64 * 32];
  int m0 = blockIdx.x * 64, n0 = blockIdx.y * 64;
  int wave = threadIdx.x >> 5, lane = threadIdx.x & 31;
  int mw = m0 + wave * 16;
  bool full = (m0 + 64 <= M) && (n0 + 64 <= Nn);
  fv8 acc[4];
#pragma unroll
  for (int t = 0; t < 4; ++t) acc[t] = (fv8){};
  int r = threadIdx.x >> 1, seg = threadIdx.x & 1;

  auto stage = [&](int buf, int k0) {
    {
      int gr = m0 + r;
      _Float16* dA = &sA[buf][0] + r * 32 + seg * 16;
      if (gr < M) {
        const _Float16* src = A + (size_t)gr * K + k0 + seg * 16;
        async_g2l_b128(dA, src);
        async_g2l_b128(dA + 8, src + 8);
      } else {
        *(hv8*)dA = (hv8){};
        *(hv8*)(dA + 8) = (hv8){};
      }
    }
    {
      int gn = n0 + r;
      _Float16* dB = &sB[buf][0] + r * 32 + seg * 16;
      if (gn < Nn) {
        const _Float16* src = Bt + (size_t)gn * K + k0 + seg * 16;
        async_g2l_b128(dB, src);
        async_g2l_b128(dB + 8, src + 8);
      } else {
        *(hv8*)dB = (hv8){};
        *(hv8*)(dB + 8) = (hv8){};
      }
    }
  };

  stage(0, 0);
  int nst = K >> 5;
  for (int s = 0; s < nst; ++s) {
    int buf = s & 1;
    if (s + 1 < nst) {
      stage(buf ^ 1, (s + 1) * 32);
      if (full) wait_asynccnt<4>();   // current stage done, next in flight
      else      wait_asynccnt<0>();
    } else {
      wait_asynccnt<0>();
    }
    __syncthreads();
    hv16 af = load_a_frag(&sA[buf][0] + (wave * 16) * 32, 32);
#pragma unroll
    for (int nt = 0; nt < 4; ++nt) {
      hv16 bf = load_b_frag(&sB[buf][0] + (nt * 16) * 32, 32);
      acc[nt] = wmma_f16(af, bf, acc[nt]);
    }
    __syncthreads();
  }

  int half = lane >> 4, colb = lane & 15;
#pragma unroll
  for (int nt = 0; nt < 4; ++nt) {
#pragma unroll
    for (int rr = 0; rr < 8; ++rr) {
      int row = mw + rr + 8 * half;
      int col = n0 + nt * 16 + colb;
      if (row >= M || col >= Nn) continue;
      float v = acc[nt][rr];
      if (bias) v += bias[col];
      size_t o = (size_t)row * Nn + col;
      if (EPI == 0) {
        outf[o] = v;
      } else if (EPI == 1) {
        outh[o] = (_Float16)v;
      } else if (EPI == 3) {
        outh[o] = (_Float16)(siluf((float)auxh[o]) * v);
      } else if (EPI == 4) {
        size_t ro = (resTransN > 0)
                        ? (size_t)((row % resTransN) * resTransN + row / resTransN) * Nn + col
                        : o;
        outf[ro] += v;
      } else if (EPI == 5) {
        size_t ro = (resTransN > 0)
                        ? (size_t)((row % resTransN) * resTransN + row / resTransN) * Nn + col
                        : o;
        outf[ro] += aux[o] * sigf(v);
      } else if (EPI == 7) {
        // tri-mul: ab = p * sigmoid(g) * pair_mask; scatter to channel-major
        float w = aux[o] * sigf(v) * pm[row];
        int d = col & 127, hb = col >> 7;
        int r0 = row >> 8, r1 = row & 255;
        size_t idx = (size_t)d * NN + (scat_trans ? (size_t)(r1 * NTOK + r0)
                                                  : (size_t)(r0 * NTOK + r1));
        scat[(size_t)hb * 128 * NN + idx] = (_Float16)w;
      }
    }
  }
}

// ---------------------------------------------------------------------------
// Triangle einsum: t[i,j,d] = sum_k A_d[i,k] * B_d[j,k]  (f16 [d][256][256]).
// Block = (d, i-tile, 64-wide j group), 4 waves. TDM stages the A panel
// (16x256) and B panel (64x256) into LDS; each wave then runs 8 WMMAs.
// Fallback (no TDM builtin): direct global-fragment loads, same launch shape.
// ---------------------------------------------------------------------------
__global__ __launch_bounds__(128) void k_tri_einsum(const _Float16* __restrict__ Ah,
                                                    const _Float16* __restrict__ Bh,
                                                    float* __restrict__ t) {
  int d = blockIdx.z;
  int i0 = blockIdx.x * 16;
  int jg = blockIdx.y * 64;
  int wave = threadIdx.x >> 5, lane = threadIdx.x & 31;
  fv8 acc = (fv8){};
#if HAVE_TDM
  __shared__ __align__(16) _Float16 sA[16 * 256];
  __shared__ __align__(16) _Float16 sB[64 * 256];
  if (threadIdx.x == 0) {
    tdm_load_2d(sA, Ah + (size_t)d * NN + (size_t)i0 * NTOK, 16, 256);
    tdm_load_2d(sB, Bh + (size_t)d * NN + (size_t)jg * NTOK, 64, 256);
  }
  wait_tensorcnt0();
  __syncthreads();
  for (int k0 = 0; k0 < NTOK; k0 += 32) {
    hv16 af = load_a_frag(sA + k0, 256);
    hv16 bf = load_b_frag(sB + (size_t)(wave * 16) * 256 + k0, 256);
    acc = wmma_f16(af, bf, acc);
  }
#else
  const _Float16* Ad = Ah + (size_t)d * NN + (size_t)i0 * NTOK;
  const _Float16* Bd = Bh + (size_t)d * NN + (size_t)(jg + wave * 16) * NTOK;
  for (int k0 = 0; k0 < NTOK; k0 += 32) {
    acc = wmma_f16(load_a_frag(Ad + k0, NTOK), load_b_frag(Bd + k0, NTOK), acc);
  }
#endif
  int half = lane >> 4, colb = lane & 15;
  int j0 = jg + wave * 16;
#pragma unroll
  for (int r = 0; r < 8; ++r) {
    int row = i0 + r + 8 * half;
    t[((size_t)row * NTOK + j0 + colb) * DZ + d] = acc[r];
  }
}

// ---------------------------------------------------------------------------
// Repack: v [i*256+j][128] (f16, heads of width 32) -> vt [i][h][w][j]
// ---------------------------------------------------------------------------
__global__ void k_pack_vt_tri(const _Float16* __restrict__ vh, _Float16* __restrict__ vt) {
  int idx = blockIdx.x * 256 + threadIdx.x;  // 256*4*32*256
  int j = idx & 255;
  int w = (idx >> 8) & 31;
  int h = (idx >> 13) & 3;
  int i = idx >> 15;
  vt[idx] = vh[((size_t)(i * NTOK + j)) * DZ + h * WT + w];
}

__global__ void k_pack_qk_pair(const _Float16* __restrict__ src, _Float16* __restrict__ dst) {
  int idx = blockIdx.x * 256 + threadIdx.x;  // 16*256*32
  if (idx >= HP * NTOK * 32) return;
  int w = idx & 31;
  int n = (idx >> 5) & 255;
  int h = idx >> 13;
  dst[idx] = (w < DP) ? src[(size_t)n * CS + h * DP + w] : (_Float16)0.0f;
}

__global__ void k_pack_vt_pair(const _Float16* __restrict__ src, _Float16* __restrict__ dst) {
  int idx = blockIdx.x * 256 + threadIdx.x;  // 16*32*256
  if (idx >= HP * 32 * NTOK) return;
  int j = idx & 255;
  int w = (idx >> 8) & 31;
  int h = idx >> 13;
  dst[idx] = (w < DP) ? src[(size_t)j * CS + h * DP + w] : (_Float16)0.0f;
}

// ---------------------------------------------------------------------------
// Flash attention for triangle attention (starting-node math; ending node is
// handled by feeding transposed z). One wave handles 16 query rows of one
// (i, h); online softmax over K=256 in chunks of 32; P goes through LDS to
// re-fragment (C-layout fp32 -> A-layout f16).
// ---------------------------------------------------------------------------
__global__ __launch_bounds__(128) void k_flash_tri(
    const _Float16* __restrict__ qh, const _Float16* __restrict__ kh,
    const _Float16* __restrict__ vt, const float* __restrict__ graw,
    const float* __restrict__ xb, const float* __restrict__ pm,
    _Float16* __restrict__ outh) {
  __shared__ __align__(16) _Float16 sP[4][16 * 32];
  int wave = threadIdx.x >> 5, lane = threadIdx.x & 31;
  int jt = blockIdx.x * 4 + wave;
  int j0 = jt * 16;
  int h = blockIdx.y;
  int i = blockIdx.z;
  const float scale = 0.17677669529663687f;  // 1/sqrt(32)
  hv16 qf = load_a_frag(qh + ((size_t)(i * NTOK + j0)) * DZ + h * WT, DZ);
  fv8 o0 = (fv8){}, o1 = (fv8){};
  float rowM[8], rowL[8];
#pragma unroll
  for (int r = 0; r < 8; ++r) { rowM[r] = -1e30f; rowL[r] = 0.f; }
  int half = lane >> 4, colb = lane & 15;
  for (int k0 = 0; k0 < NTOK; k0 += 32) {
    hv16 b0 = load_b_frag(kh + ((size_t)(i * NTOK + k0)) * DZ + h * WT, DZ);
    hv16 b1 = load_b_frag(kh + ((size_t)(i * NTOK + k0 + 16)) * DZ + h * WT, DZ);
    fv8 s0 = wmma_f16(qf, b0, (fv8){});
    fv8 s1 = wmma_f16(qf, b1, (fv8){});
    int c0 = k0 + colb, c1 = k0 + 16 + colb;
    float mb0 = INFB * (pm[i * NTOK + c0] - 1.f);
    float mb1 = INFB * (pm[i * NTOK + c1] - 1.f);
#pragma unroll
    for (int r = 0; r < 8; ++r) {
      int row = j0 + r + 8 * half;
      s0[r] = s0[r] * scale + xb[((size_t)row * NTOK + c0) * 4 + h] + mb0;
      s1[r] = s1[r] * scale + xb[((size_t)row * NTOK + c1) * 4 + h] + mb1;
      float mt = fmaxf(s0[r], s1[r]);
#pragma unroll
      for (int off = 1; off < 16; off <<= 1) mt = fmaxf(mt, __shfl_xor(mt, off, 16));
      float mnew = fmaxf(rowM[r], mt);
      float corr = __expf(rowM[r] - mnew);
      rowM[r] = mnew;
      float p0 = __expf(s0[r] - mnew);
      float p1 = __expf(s1[r] - mnew);
      float rs = p0 + p1;
#pragma unroll
      for (int off = 1; off < 16; off <<= 1) rs += __shfl_xor(rs, off, 16);
      rowL[r] = rowL[r] * corr + rs;
      o0[r] *= corr; o1[r] *= corr;
      int rl = r + 8 * half;
      sP[wave][rl * 32 + colb] = (_Float16)p0;
      sP[wave][rl * 32 + 16 + colb] = (_Float16)p1;
    }
    hv16 pa = load_a_frag(&sP[wave][0], 32);
    hv16 bv0 = load_b_frag(vt + (((size_t)(i * HT + h) * 32 + 0) * NTOK + k0), NTOK);
    hv16 bv1 = load_b_frag(vt + (((size_t)(i * HT + h) * 32 + 16) * NTOK + k0), NTOK);
    o0 = wmma_f16(pa, bv0, o0);
    o1 = wmma_f16(pa, bv1, o1);
  }
#pragma unroll
  for (int r = 0; r < 8; ++r) {
    float inv = 1.f / (rowL[r] + 1e-20f);
    int row = j0 + r + 8 * half;
    size_t base = (size_t)(i * NTOK + row) * DZ + h * WT;
    outh[base + colb] = (_Float16)(o0[r] * inv * sigf(graw[base + colb]));
    outh[base + 16 + colb] = (_Float16)(o1[r] * inv * sigf(graw[base + 16 + colb]));
  }
}

// ---------------------------------------------------------------------------
// Flash attention for attention_pair_bias. qp/kp [h][256][32] (D=24 padded),
// vt [h][32][256], bias zproj [q*256+key][16], gate/out in [256][384].
// ---------------------------------------------------------------------------
__global__ __launch_bounds__(128) void k_flash_pair(
    const _Float16* __restrict__ qp, const _Float16* __restrict__ kp,
    const _Float16* __restrict__ vt, const float* __restrict__ graw,
    const float* __restrict__ zp, const float* __restrict__ mask,
    _Float16* __restrict__ outh) {
  __shared__ __align__(16) _Float16 sP[4][16 * 32];
  int wave = threadIdx.x >> 5, lane = threadIdx.x & 31;
  int q0 = (blockIdx.x * 4 + wave) * 16;
  int h = blockIdx.y;
  const float scale = 0.20412414523193154f;  // 1/sqrt(24)
  hv16 qf = load_a_frag(qp + ((size_t)(h * NTOK + q0)) * 32, 32);
  fv8 o0 = (fv8){}, o1 = (fv8){};
  float rowM[8], rowL[8];
#pragma unroll
  for (int r = 0; r < 8; ++r) { rowM[r] = -1e30f; rowL[r] = 0.f; }
  int half = lane >> 4, colb = lane & 15;
  for (int k0 = 0; k0 < NTOK; k0 += 32) {
    hv16 b0 = load_b_frag(kp + ((size_t)(h * NTOK + k0)) * 32, 32);
    hv16 b1 = load_b_frag(kp + ((size_t)(h * NTOK + k0 + 16)) * 32, 32);
    fv8 s0 = wmma_f16(qf, b0, (fv8){});
    fv8 s1 = wmma_f16(qf, b1, (fv8){});
    int c0 = k0 + colb, c1 = k0 + 16 + colb;
    float mb0 = INFB * (mask[c0] - 1.f);
    float mb1 = INFB * (mask[c1] - 1.f);
#pragma unroll
    for (int r = 0; r < 8; ++r) {
      int qrow = q0 + r + 8 * half;
      s0[r] = s0[r] * scale + zp[((size_t)qrow * NTOK + c0) * HP + h] + mb0;
      s1[r] = s1[r] * scale + zp[((size_t)qrow * NTOK + c1) * HP + h] + mb1;
      float mt = fmaxf(s0[r], s1[r]);
#pragma unroll
      for (int off = 1; off < 16; off <<= 1) mt = fmaxf(mt, __shfl_xor(mt, off, 16));
      float mnew = fmaxf(rowM[r], mt);
      float corr = __expf(rowM[r] - mnew);
      rowM[r] = mnew;
      float p0 = __expf(s0[r] - mnew);
      float p1 = __expf(s1[r] - mnew);
      float rs = p0 + p1;
#pragma unroll
      for (int off = 1; off < 16; off <<= 1) rs += __shfl_xor(rs, off, 16);
      rowL[r] = rowL[r] * corr + rs;
      o0[r] *= corr; o1[r] *= corr;
      int rl = r + 8 * half;
      sP[wave][rl * 32 + colb] = (_Float16)p0;
      sP[wave][rl * 32 + 16 + colb] = (_Float16)p1;
    }
    hv16 pa = load_a_frag(&sP[wave][0], 32);
    hv16 bv0 = load_b_frag(vt + (((size_t)h * 32 + 0) * NTOK + k0), NTOK);
    hv16 bv1 = load_b_frag(vt + (((size_t)h * 32 + 16) * NTOK + k0), NTOK);
    o0 = wmma_f16(pa, bv0, o0);
    o1 = wmma_f16(pa, bv1, o1);
  }
#pragma unroll
  for (int r = 0; r < 8; ++r) {
    float inv = 1.f / (rowL[r] + 1e-20f);
    int qrow = q0 + r + 8 * half;
    size_t base = (size_t)qrow * CS + h * DP;
    outh[base + colb] = (_Float16)(o0[r] * inv * sigf(graw[base + colb]));
    int w1 = 16 + colb;
    if (w1 < DP) outh[base + w1] = (_Float16)(o1[r] * inv * sigf(graw[base + w1]));
  }
}

// ===========================================================================
// Host orchestration
// ===========================================================================
extern "C" void kernel_launch(void* const* d_in, const int* in_sizes, int n_in,
                              void* d_out, int out_size, void* d_ws, size_t ws_size,
                              hipStream_t stream) {
  (void)in_sizes; (void)n_in; (void)out_size; (void)ws_size;
  // ---- input index map (dict-insertion-order flatten of setup_inputs) ----
  const float* in_s    = (const float*)d_in[0];
  const float* in_z    = (const float*)d_in[1];
  const float* in_mask = (const float*)d_in[2];
  // tri_mul_out: 3..10  tri_mul_in: 11..18
  // tri_att_start: 19..26  tri_att_end: 27..34
  // trans_z: 35..39  trans_s: 40..44  attn: 45..55
  auto F = [&](int i) { return (const float*)d_in[i]; };

  // ---- workspace bump allocator (deterministic) ----
  char* wsp = (char*)d_ws;
  auto alloc = [&](size_t bytes) {
    char* p = wsp;
    wsp += (bytes + 255) & ~(size_t)255;
    return p;
  };
  _Float16* WT_ = (_Float16*)alloc(8u << 20);
  float*    PM  = (float*)alloc((size_t)NN * 4);
  _Float16* H1  = (_Float16*)alloc(16u << 20);
  _Float16* H2  = (_Float16*)alloc(64u << 20);
  _Float16* H3  = (_Float16*)alloc(64u << 20);
  float*    F1  = (float*)alloc(64u << 20);
  float*    F2  = (float*)alloc(80u << 20);

  // ---- working output copies (s then z in d_out) ----
  float* s_work = (float*)d_out;
  float* z_work = (float*)d_out + NTOK * CS;
  k_copy_f32<<<(NTOK * CS + 255) / 256, 256, 0, stream>>>(in_s, s_work, NTOK * CS);
  k_copy_f32<<<(NN * DZ + 255) / 256, 256, 0, stream>>>(in_z, z_work, NN * DZ);
  k_pm<<<NTOK, NTOK, 0, stream>>>(in_mask, PM);

  // ---- convert all weights fp32[K][N] -> f16[N][K] ----
  _Float16* wcur = WT_;
  auto cvt = [&](int idx, int K, int N) {
    _Float16* dst = wcur;
    wcur += (size_t)K * N;
    k_wt<<<(K * N + 255) / 256, 256, 0, stream>>>(F(idx), dst, K, N);
    return (const _Float16*)dst;
  };
  // tri_mul weights
  const _Float16* W_tmo_pin = cvt(5, DZ, 2 * DZ);
  const _Float16* W_tmo_gin = cvt(6, DZ, 2 * DZ);
  const _Float16* W_tmo_pout = cvt(9, DZ, DZ);
  const _Float16* W_tmo_gout = cvt(10, DZ, DZ);
  const _Float16* W_tmi_pin = cvt(13, DZ, 2 * DZ);
  const _Float16* W_tmi_gin = cvt(14, DZ, 2 * DZ);
  const _Float16* W_tmi_pout = cvt(17, DZ, DZ);
  const _Float16* W_tmi_gout = cvt(18, DZ, DZ);
  // tri_att weights
  const _Float16* W_tas_bias = cvt(21, DZ, HT);
  const _Float16* W_tas_q = cvt(22, DZ, DZ);
  const _Float16* W_tas_k = cvt(23, DZ, DZ);
  const _Float16* W_tas_v = cvt(24, DZ, DZ);
  const _Float16* W_tas_g = cvt(25, DZ, DZ);
  const _Float16* W_tas_o = cvt(26, DZ, DZ);
  const _Float16* W_tae_bias = cvt(29, DZ, HT);
  const _Float16* W_tae_q = cvt(30, DZ, DZ);
  const _Float16* W_tae_k = cvt(31, DZ, DZ);
  const _Float16* W_tae_v = cvt(32, DZ, DZ);
  const _Float16* W_tae_g = cvt(33, DZ, DZ);
  const _Float16* W_tae_o = cvt(34, DZ, DZ);
  // transitions
  const _Float16* W_tz_fc1 = cvt(37, DZ, 4 * DZ);
  const _Float16* W_tz_fc2 = cvt(38, DZ, 4 * DZ);
  const _Float16* W_tz_fc3 = cvt(39, 4 * DZ, DZ);
  const _Float16* W_ts_fc1 = cvt(42, CS, 4 * CS);
  const _Float16* W_ts_fc2 = cvt(43, CS, 4 * CS);
  const _Float16* W_ts_fc3 = cvt(44, 4 * CS, CS);
  // attention pair bias
  const _Float16* W_at_q = cvt(47, CS, CS);
  const _Float16* W_at_k = cvt(49, CS, CS);
  const _Float16* W_at_v = cvt(50, CS, CS);
  const _Float16* W_at_g = cvt(51, CS, CS);
  const _Float16* W_at_zp = cvt(54, DZ, HP);
  const _Float16* W_at_o = cvt(55, CS, CS);

  dim3 blk(128);
  auto gemm_grid = [](int M, int Nn) { return dim3((M + 63) / 64, (Nn + 63) / 64); };

  // ======================= tri_mul (outgoing then incoming) ================
  auto tri_mul = [&](int bi, const _Float16* Wpin, const _Float16* Wgin,
                     const _Float16* Wpout, const _Float16* Wgout, int trans) {
    // x = ln(z)
    k_ln<<<NN, 128, 0, stream>>>(z_work, F(bi), F(bi + 1), DZ, 0, H1);
    // p = x @ p_in  (fp32 aux)
    k_gemm<0><<<gemm_grid(NN, 2 * DZ), blk, 0, stream>>>(
        H1, Wpin, NN, 2 * DZ, DZ, nullptr, nullptr, nullptr, F1, nullptr,
        nullptr, nullptr, 0, 0);
    // ab = p*sigmoid(x@g_in)*pm -> channel-major f16 scatter (a=H2, b=H2+128*NN)
    k_gemm<7><<<gemm_grid(NN, 2 * DZ), blk, 0, stream>>>(
        H1, Wgin, NN, 2 * DZ, DZ, nullptr, F1, nullptr, nullptr, nullptr, PM,
        H2, trans, 0);
    // t[i,j,d] = sum_k a b
    k_tri_einsum<<<dim3(16, 4, DZ), blk, 0, stream>>>(
        H2, H2 + (size_t)DZ * NN, F2);
    // tn = ln(t)
    k_ln<<<NN, 128, 0, stream>>>(F2, F(bi + 4), F(bi + 5), DZ, 0, H3);
    // o1 = tn @ p_out (fp32)
    float* o1 = F2 + (size_t)NN * DZ;
    k_gemm<0><<<gemm_grid(NN, DZ), blk, 0, stream>>>(
        H3, Wpout, NN, DZ, DZ, nullptr, nullptr, nullptr, o1, nullptr, nullptr,
        nullptr, 0, 0);
    // z += o1 * sigmoid(x @ g_out)
    k_gemm<5><<<gemm_grid(NN, DZ), blk, 0, stream>>>(
        H1, Wgout, NN, DZ, DZ, nullptr, o1, nullptr, z_work, nullptr, nullptr,
        nullptr, 0, 0);
  };
  tri_mul(3, W_tmo_pin, W_tmo_gin, W_tmo_pout, W_tmo_gout, 0);
  tri_mul(11, W_tmi_pin, W_tmi_gin, W_tmi_pout, W_tmi_gout, 1);

  // ======================= tri_att (starting then ending) ==================
  auto tri_att = [&](int bi, const _Float16* Wb, const _Float16* Wq,
                     const _Float16* Wk, const _Float16* Wv, const _Float16* Wg,
                     const _Float16* Wo, int ending) {
    k_ln<<<NN, 128, 0, stream>>>(z_work, F(bi), F(bi + 1), DZ, ending ? NTOK : 0, H1);
    // tri bias xb [NN][4]
    k_gemm<0><<<gemm_grid(NN, HT), blk, 0, stream>>>(
        H1, Wb, NN, HT, DZ, nullptr, nullptr, nullptr, F2, nullptr, nullptr,
        nullptr, 0, 0);
    _Float16* qh = H2;
    _Float16* kh = H2 + (size_t)NN * DZ;
    _Float16* vh = H3;
    _Float16* vt = H3 + (size_t)NN * DZ;
    _Float16* oa = H3 + 2 * (size_t)NN * DZ;
    k_gemm<1><<<gemm_grid(NN, DZ), blk, 0, stream>>>(
        H1, Wq, NN, DZ, DZ, nullptr, nullptr, nullptr, nullptr, qh, nullptr,
        nullptr, 0, 0);
    k_gemm<1><<<gemm_grid(NN, DZ), blk, 0, stream>>>(
        H1, Wk, NN, DZ, DZ, nullptr, nullptr, nullptr, nullptr, kh, nullptr,
        nullptr, 0, 0);
    k_gemm<1><<<gemm_grid(NN, DZ), blk, 0, stream>>>(
        H1, Wv, NN, DZ, DZ, nullptr, nullptr, nullptr, nullptr, vh, nullptr,
        nullptr, 0, 0);
    k_gemm<0><<<gemm_grid(NN, DZ), blk, 0, stream>>>(
        H1, Wg, NN, DZ, DZ, nullptr, nullptr, nullptr, F1, nullptr, nullptr,
        nullptr, 0, 0);
    k_pack_vt_tri<<<(NN * DZ) / 256, 256, 0, stream>>>(vh, vt);
    k_flash_tri<<<dim3(4, HT, NTOK), blk, 0, stream>>>(qh, kh, vt, F1, F2, PM, oa);
    // z += (g*o) @ Wo   (transposed residual for ending node)
    k_gemm<4><<<gemm_grid(NN, DZ), blk, 0, stream>>>(
        oa, Wo, NN, DZ, DZ, nullptr, nullptr, nullptr, z_work, nullptr, nullptr,
        nullptr, 0, ending ? NTOK : 0);
  };
  tri_att(19, W_tas_bias, W_tas_q, W_tas_k, W_tas_v, W_tas_g, W_tas_o, 0);
  tri_att(27, W_tae_bias, W_tae_q, W_tae_k, W_tae_v, W_tae_g, W_tae_o, 1);

  // ======================= transition(z) ===================================
  auto transition = [&](float* x_work, int rows, int dim, int bi,
                        const _Float16* Wf1, const _Float16* Wf2,
                        const _Float16* Wf3) {
    k_ln<<<rows, 128, 0, stream>>>(x_work, F(bi), F(bi + 1), dim, 0, H1);
    k_gemm<1><<<gemm_grid(rows, 4 * dim), blk, 0, stream>>>(
        H1, Wf1, rows, 4 * dim, dim, nullptr, nullptr, nullptr, nullptr, H2,
        nullptr, nullptr, 0, 0);
    k_gemm<3><<<gemm_grid(rows, 4 * dim), blk, 0, stream>>>(
        H1, Wf2, rows, 4 * dim, dim, nullptr, nullptr, H2, nullptr, H3, nullptr,
        nullptr, 0, 0);
    k_gemm<4><<<gemm_grid(rows, dim), blk, 0, stream>>>(
        H3, Wf3, rows, dim, 4 * dim, nullptr, nullptr, nullptr, x_work, nullptr,
        nullptr, nullptr, 0, 0);
  };
  transition(z_work, NN, DZ, 35, W_tz_fc1, W_tz_fc2, W_tz_fc3);

  // ======================= attention_pair_bias =============================
  {
    _Float16* sn = H2;                             // [256][384]
    _Float16* qh = H2 + (size_t)NTOK * CS;         // [256][384]
    _Float16* kh = H2 + 2 * (size_t)NTOK * CS;
    _Float16* vh = H2 + 3 * (size_t)NTOK * CS;
    _Float16* qp = H3;                             // [16][256][32]
    _Float16* kp = H3 + (size_t)HP * NTOK * 32;
    _Float16* vt = H3 + 2 * (size_t)HP * NTOK * 32;
    _Float16* os = H3 + 3 * (size_t)HP * NTOK * 32;  // [256][384]
    k_ln<<<NTOK, 128, 0, stream>>>(s_work, F(45), F(46), CS, 0, sn);
    k_gemm<1><<<gemm_grid(NTOK, CS), blk, 0, stream>>>(
        sn, W_at_q, NTOK, CS, CS, F(48), nullptr, nullptr, nullptr, qh, nullptr,
        nullptr, 0, 0);
    k_gemm<1><<<gemm_grid(NTOK, CS), blk, 0, stream>>>(
        sn, W_at_k, NTOK, CS, CS, nullptr, nullptr, nullptr, nullptr, kh,
        nullptr, nullptr, 0, 0);
    k_gemm<1><<<gemm_grid(NTOK, CS), blk, 0, stream>>>(
        sn, W_at_v, NTOK, CS, CS, nullptr, nullptr, nullptr, nullptr, vh,
        nullptr, nullptr, 0, 0);
    k_gemm<0><<<gemm_grid(NTOK, CS), blk, 0, stream>>>(
        sn, W_at_g, NTOK, CS, CS, nullptr, nullptr, nullptr, F1, nullptr,
        nullptr, nullptr, 0, 0);
    // bias = ln(z) @ z_proj  -> F2 [NN][16]
    k_ln<<<NN, 128, 0, stream>>>(z_work, F(52), F(53), DZ, 0, H1);
    k_gemm<0><<<gemm_grid(NN, HP), blk, 0, stream>>>(
        H1, W_at_zp, NN, HP, DZ, nullptr, nullptr, nullptr, F2, nullptr,
        nullptr, nullptr, 0, 0);
    k_pack_qk_pair<<<(HP * NTOK * 32 + 255) / 256, 256, 0, stream>>>(qh, qp);
    k_pack_qk_pair<<<(HP * NTOK * 32 + 255) / 256, 256, 0, stream>>>(kh, kp);
    k_pack_vt_pair<<<(HP * 32 * NTOK + 255) / 256, 256, 0, stream>>>(vh, vt);
    k_flash_pair<<<dim3(4, HP, 1), blk, 0, stream>>>(qp, kp, vt, F1, F2, in_mask, os);
    k_gemm<4><<<gemm_grid(NTOK, CS), blk, 0, stream>>>(
        os, W_at_o, NTOK, CS, CS, nullptr, nullptr, nullptr, s_work, nullptr,
        nullptr, nullptr, 0, 0);
  }

  // ======================= transition(s) ===================================
  transition(s_work, NTOK, CS, 40, W_ts_fc1, W_ts_fc2, W_ts_fc3);
}
